// DualGraphConvolution_29283087024207
// MI455X (gfx1250) — matrix operand order repaired and checked by reading, not compiled, not run on hardware
//
#include <hip/hip_runtime.h>

// ---------------------------------------------------------------------------
// DualGraphConvolution on MI455X (gfx1250, wave32, WMMA + TDM)
// N = 8192, D_IN = D_OUT = 512
// ---------------------------------------------------------------------------

#define GN 8192
#define GD 512
#define NEG_BIG_F (-3.0e38f)

#define PROW 520                      // padded LDS row stride (ushorts): 1024B + 16B pad
#define PANEL_USH (32 * PROW)         // one h-panel in ushorts
#define PANEL_BYTES (PANEL_USH * 2)   // 33280 B
#define FLASH_LDS_BYTES (2 * PANEL_BYTES + 8 * 16 * 32 * 4 + 16 * 32 * 4)

typedef __attribute__((ext_vector_type(16))) __bf16 bf16x16;
typedef __attribute__((ext_vector_type(8)))  float  f32x8;
typedef __attribute__((ext_vector_type(4)))  unsigned int u32x4;
typedef __attribute__((ext_vector_type(8)))  unsigned int u32x8;

union B16U {
    uint4          q[2];
    unsigned short u[16];
    bf16x16        v;
};

// float -> bf16 (round-to-nearest-even), storage as ushort
static __device__ __forceinline__ unsigned short f2bf(float f) {
    unsigned u = __float_as_uint(f);
    unsigned r = (u + 0x7FFFu + ((u >> 16) & 1u)) >> 16;
    return (unsigned short)r;
}

// B fragment: 16 contiguous bf16 (32 bytes)
static __device__ __forceinline__ bf16x16 ld_bfrag(const unsigned short* p) {
    B16U t;
    t.q[0] = *reinterpret_cast<const uint4*>(p);
    t.q[1] = *reinterpret_cast<const uint4*>(p + 8);
    return t.v;
}

// A fragment: 8 bf16 at p, 8 bf16 at p+16 (K chunks k0..k0+7, k0+16..k0+23)
static __device__ __forceinline__ bf16x16 ld_afrag(const unsigned short* p) {
    B16U t;
    t.q[0] = *reinterpret_cast<const uint4*>(p);
    t.q[1] = *reinterpret_cast<const uint4*>(p + 16);
    return t.v;
}

static __device__ __forceinline__ f32x8 wmma_bf16(bf16x16 a, bf16x16 b, f32x8 c) {
    return __builtin_amdgcn_wmma_f32_16x16x32_bf16(
        /*neg_a=*/false, a, /*neg_b=*/false, b,
        /*c_mod=*/(short)0, c, /*reuse_a=*/false, /*reuse_b=*/false);
}

// ---------------------------------------------------------------------------
// Tensor Data Mover: async load of a 32-row x 512-bf16 panel into LDS with
// 16B/row padding (bank rotation).  D# per CDNA5 ISA.
// ---------------------------------------------------------------------------
static __device__ __forceinline__ void tdm_load_panel(const unsigned short* gsrc,
                                                      unsigned lds_byte_addr) {
    unsigned long long ga = (unsigned long long)(uintptr_t)gsrc;
    u32x4 g0;
    g0[0] = 1u;                                               // count=1, user mode
    g0[1] = lds_byte_addr;                                    // lds_addr
    g0[2] = (unsigned)ga;                                     // global_addr[31:0]
    g0[3] = (unsigned)((ga >> 32) & 0x1FFFFFFu) | (2u << 30); // addr[56:32] | type=2
    u32x8 g1;
    g1[0] = (1u << 16)        // data_size = 2 bytes
          | (1u << 20)        // pad_enable
          | (7u << 22)        // pad_interval: 256 dwords (= one 1024B row)
          | (3u << 25);       // pad_amount: 4 dwords (16B)
    g1[1] = (GD & 0xFFFFu) << 16;     // tensor_dim0 = 512   (bits 79:48)
    g1[2] = (32u << 16);              // tensor_dim1 = 32    (bits 111:80)
    g1[3] = ((unsigned)GD << 16);     // tile_dim0 = 512     (bits 127:112)
    g1[4] = 32u;                      // tile_dim1 = 32, tile_dim2 = 0
    g1[5] = (unsigned)GD;             // tensor_dim0_stride = 512
    g1[6] = 0u;
    g1[7] = 0u;
    asm volatile("tensor_load_to_lds %0, %1" :: "s"(g0), "s"(g1) : "memory");
}

// ---------------------------------------------------------------------------
// Small prep kernels
// ---------------------------------------------------------------------------

__global__ __launch_bounds__(256) void k_cvt_bf16(const float* __restrict__ in,
                                                  unsigned short* __restrict__ out,
                                                  int n4) {
    int i = blockIdx.x * 256 + threadIdx.x;
    if (i >= n4) return;
    float4 v = *reinterpret_cast<const float4*>(in + (size_t)i * 4);
    uint2 o;
    o.x = (unsigned)f2bf(v.x) | ((unsigned)f2bf(v.y) << 16);
    o.y = (unsigned)f2bf(v.z) | ((unsigned)f2bf(v.w) << 16);
    *reinterpret_cast<uint2*>(out + (size_t)i * 4) = o;
}

__global__ __launch_bounds__(256) void k_transpose_cvt(const float* __restrict__ in,
                                                       unsigned short* __restrict__ out) {
    int idx = blockIdx.x * 256 + threadIdx.x;   // 0 .. 512*512-1
    int j = idx >> 9;
    int k = idx & 511;
    out[idx] = f2bf(in[k * GD + j]);
}

__global__ __launch_bounds__(256) void k_node_scores(const float* __restrict__ x,
                                                     const float* __restrict__ nw,
                                                     float* __restrict__ ns) {
    int lane = threadIdx.x & 31;
    int w    = threadIdx.x >> 5;
    int row  = blockIdx.x * 8 + w;
    const float* xr = x + (size_t)row * GD + lane * 16;
    const float* wr = nw + lane * 16;
    float s = 0.f;
#pragma unroll
    for (int i = 0; i < 4; ++i) {
        float4 a = *reinterpret_cast<const float4*>(xr + i * 4);
        float4 b = *reinterpret_cast<const float4*>(wr + i * 4);
        s += a.x * b.x + a.y * b.y + a.z * b.z + a.w * b.w;
    }
#pragma unroll
    for (int o = 16; o > 0; o >>= 1) s += __shfl_xor(s, o, 32);
    if (lane == 0) ns[row] = s;
}

__global__ __launch_bounds__(256) void k_node_softmax(const float* __restrict__ ns,
                                                      float* __restrict__ na) {
    __shared__ float red[256];
    int t = threadIdx.x;
    float mx = NEG_BIG_F;
    for (int i = t; i < GN; i += 256) mx = fmaxf(mx, ns[i]);
    red[t] = mx;
    __syncthreads();
    for (int o = 128; o > 0; o >>= 1) {
        if (t < o) red[t] = fmaxf(red[t], red[t + o]);
        __syncthreads();
    }
    mx = red[0];
    __syncthreads();
    float sm = 0.f;
    for (int i = t; i < GN; i += 256) sm += __expf(ns[i] - mx);
    red[t] = sm;
    __syncthreads();
    for (int o = 128; o > 0; o >>= 1) {
        if (t < o) red[t] += red[t + o];
        __syncthreads();
    }
    float inv = 1.0f / red[0];
    for (int i = t; i < GN; i += 256) na[i] = __expf(ns[i] - mx) * inv;
}

// ---------------------------------------------------------------------------
// WMMA GEMM:  C[8192 x 512] = xbf @ W   (W given transposed: WT[512][512] bf16)
// wave = 16 rows x 64 cols; software-pipelined K loop.
// ---------------------------------------------------------------------------
__global__ __launch_bounds__(256) void k_gemm_xw(const unsigned short* __restrict__ xbf,
                                                 const unsigned short* __restrict__ WT,
                                                 unsigned short* __restrict__ out,
                                                 int transpose_out) {
    int lane = threadIdx.x & 31;
    int w    = threadIdx.x >> 5;
    int m    = lane & 15;
    int hi   = lane >> 4;
    int rowbase = blockIdx.x * 16;
    int colbase = w * 64;

    f32x8 acc0 = {}, acc1 = {}, acc2 = {}, acc3 = {};
    const unsigned short* xr = xbf + (size_t)(rowbase + m) * GD + hi * 16;
    const unsigned short* b0 = WT + (size_t)(colbase + 0 * 16 + m) * GD + hi * 16;
    const unsigned short* b1 = WT + (size_t)(colbase + 1 * 16 + m) * GD + hi * 16;
    const unsigned short* b2 = WT + (size_t)(colbase + 2 * 16 + m) * GD + hi * 16;
    const unsigned short* b3 = WT + (size_t)(colbase + 3 * 16 + m) * GD + hi * 16;

    bf16x16 A  = ld_bfrag(xr);
    bf16x16 B0 = ld_bfrag(b0);
    bf16x16 B1 = ld_bfrag(b1);
    bf16x16 B2 = ld_bfrag(b2);
    bf16x16 B3 = ld_bfrag(b3);

#pragma unroll
    for (int ks = 0; ks < GD; ks += 32) {
        bf16x16 An = A, C0 = B0, C1 = B1, C2 = B2, C3 = B3;
        if (ks + 32 < GD) {          // compile-time after full unroll
            An = ld_bfrag(xr + ks + 32);
            C0 = ld_bfrag(b0 + ks + 32);
            C1 = ld_bfrag(b1 + ks + 32);
            C2 = ld_bfrag(b2 + ks + 32);
            C3 = ld_bfrag(b3 + ks + 32);
        }
        acc0 = wmma_bf16(A, B0, acc0);
        acc1 = wmma_bf16(A, B1, acc1);
        acc2 = wmma_bf16(A, B2, acc2);
        acc3 = wmma_bf16(A, B3, acc3);
        A = An; B0 = C0; B1 = C1; B2 = C2; B3 = C3;
    }

#pragma unroll
    for (int r = 0; r < 8; ++r) {
        int row = rowbase + r + hi * 8;
#pragma unroll
        for (int c = 0; c < 4; ++c) {
            float v = (c == 0) ? acc0[r] : (c == 1) ? acc1[r] : (c == 2) ? acc2[r] : acc3[r];
            int col = colbase + c * 16 + m;
            if (transpose_out)
                out[(size_t)col * GN + row] = f2bf(v);
            else
                out[(size_t)row * GD + col] = f2bf(v);
        }
    }
}

// ---------------------------------------------------------------------------
// Fused masked-softmax attention (flash style), TDM-fed:
//   out[i,:] = na[i]/l_i * sum_j [adj_ij>0] * exp(h_i.h_j - m_i) * supp[j,:] + bias
// Workgroup: 8 waves, 16 shared rows. Per 32-col j-block:
//   wave0 TDM-prefetches the NEXT h_j panel into a double-buffered LDS panel
//   (padded rows -> bank rotation); phase1 does the K-split hh^T from LDS;
//   deterministic LDS tree reduce; phase2 online softmax + P @ suppT.
// Dynamic LDS layout: [2 x h-panel 33280B][Spart 16KB][Sfin 2KB]
// ---------------------------------------------------------------------------
__global__ __launch_bounds__(256) void k_flash(const float* __restrict__ adj,
                                               const unsigned short* __restrict__ h,
                                               const unsigned short* __restrict__ suppT,
                                               const float* __restrict__ na,
                                               const float* __restrict__ bias,
                                               float* __restrict__ out) {
    extern __shared__ char smem[];
    unsigned short* hpan  = (unsigned short*)smem;                       // 2 panels
    float*          Spart = (float*)(smem + 2 * PANEL_BYTES);            // [8][512]
    float*          Sfin  = (float*)(smem + 2 * PANEL_BYTES + 8 * 512 * 4);

    int lane = threadIdx.x & 31;
    int w    = threadIdx.x >> 5;
    int m    = lane & 15;
    int hi   = lane >> 4;
    int k0   = hi * 16;
    int rowbase = blockIdx.x * 16;
    int colbase = w * 64;
    int wk      = w * 64;          // this wave's K-chunk of the score GEMM

    // Preload A fragments of h for the 16 shared rows, K = wk..wk+63
    const unsigned short* hrow = h + (size_t)(rowbase + m) * GD + wk;
    bf16x16 Ah0 = ld_bfrag(hrow + 0  + k0);
    bf16x16 Ah1 = ld_bfrag(hrow + 32 + k0);

    const unsigned short* sp0 = suppT + (size_t)(colbase + 0 * 16 + m) * GN + hi * 16;
    const unsigned short* sp1 = suppT + (size_t)(colbase + 1 * 16 + m) * GN + hi * 16;
    const unsigned short* sp2 = suppT + (size_t)(colbase + 2 * 16 + m) * GN + hi * 16;
    const unsigned short* sp3 = suppT + (size_t)(colbase + 3 * 16 + m) * GN + hi * 16;
    const float* adjrow = adj + (size_t)(rowbase + m) * GN + k0;

    f32x8 O0 = {}, O1 = {}, O2 = {}, O3 = {};
    float mrun = NEG_BIG_F;
    float lrun = 0.f;

    // prologue: blocking TDM fill of panel 0
    if (w == 0) {
        tdm_load_panel(h, (unsigned)(uintptr_t)hpan);
        __builtin_amdgcn_s_wait_tensorcnt(0);
    }
    __syncthreads();

    for (int jb = 0; jb < GN; jb += 32) {
        int buf  = (jb >> 5) & 1;
        int nbuf = buf ^ 1;

        // async prefetch of next panel (runs under this whole iteration)
        if (w == 0) {
            int jn = (jb + 32 < GN) ? jb + 32 : jb;
            tdm_load_panel(h + (size_t)jn * GD,
                           (unsigned)(uintptr_t)(hpan + nbuf * PANEL_USH));
        }

        // -------- phase 2 data: issue now, consumed after 2 barriers -------
        float av[16];
        *reinterpret_cast<float4*>(&av[0])  = *reinterpret_cast<const float4*>(adjrow + jb);
        *reinterpret_cast<float4*>(&av[4])  = *reinterpret_cast<const float4*>(adjrow + jb + 4);
        *reinterpret_cast<float4*>(&av[8])  = *reinterpret_cast<const float4*>(adjrow + jb + 8);
        *reinterpret_cast<float4*>(&av[12]) = *reinterpret_cast<const float4*>(adjrow + jb + 12);
        bf16x16 Bs0 = ld_bfrag(sp0 + jb);
        bf16x16 Bs1 = ld_bfrag(sp1 + jb);
        bf16x16 Bs2 = ld_bfrag(sp2 + jb);
        bf16x16 Bs3 = ld_bfrag(sp3 + jb);
        {
            int jn = (jb + 32 < GN) ? jb + 32 : jb;
            __builtin_prefetch(adjrow + jn, 0, 3);    // global_prefetch_b8
        }

        // -------- phase 1: partial scores from the LDS panel ---------------
        const unsigned short* pb  = hpan + buf * PANEL_USH + wk + hi * 16;
        const unsigned short* hj0 = pb + (0  + m) * PROW;
        const unsigned short* hj1 = pb + (16 + m) * PROW;
        bf16x16 Bh0a = ld_bfrag(hj0);
        bf16x16 Bh0b = ld_bfrag(hj0 + 32);
        bf16x16 Bh1a = ld_bfrag(hj1);
        bf16x16 Bh1b = ld_bfrag(hj1 + 32);

        f32x8 acc0 = {}, acc1 = {};
        acc0 = wmma_bf16(Ah0, Bh0a, acc0);
        acc0 = wmma_bf16(Ah1, Bh0b, acc0);
        acc1 = wmma_bf16(Ah0, Bh1a, acc1);
        acc1 = wmma_bf16(Ah1, Bh1b, acc1);
#pragma unroll
        for (int r = 0; r < 8; ++r) {
            Spart[w * 512 + (r + hi * 8) * 32 + 0  + m] = acc0[r];
            Spart[w * 512 + (r + hi * 8) * 32 + 16 + m] = acc1[r];
        }
        __syncthreads();

        // -------- deterministic reduction of 8 partials --------------------
        {
            int e = threadIdx.x * 2;
            float s0 = 0.f, s1 = 0.f;
#pragma unroll
            for (int p = 0; p < 8; ++p) { s0 += Spart[p * 512 + e]; s1 += Spart[p * 512 + e + 1]; }
            Sfin[e] = s0;
            Sfin[e + 1] = s1;
        }
        __syncthreads();

        // -------- phase 2: online softmax + P @ suppT ----------------------
        float sv[16];
        *reinterpret_cast<float4*>(&sv[0])  = *reinterpret_cast<float4*>(&Sfin[m * 32 + k0]);
        *reinterpret_cast<float4*>(&sv[4])  = *reinterpret_cast<float4*>(&Sfin[m * 32 + k0 + 4]);
        *reinterpret_cast<float4*>(&sv[8])  = *reinterpret_cast<float4*>(&Sfin[m * 32 + k0 + 8]);
        *reinterpret_cast<float4*>(&sv[12]) = *reinterpret_cast<float4*>(&Sfin[m * 32 + k0 + 12]);

        float mt = NEG_BIG_F;
#pragma unroll
        for (int i = 0; i < 16; ++i) {
            float s = (av[i] > 0.f) ? sv[i] : NEG_BIG_F;
            mt = fmaxf(mt, s);
        }
        mt = fmaxf(mt, __shfl_xor(mt, 16, 32));       // row spans lanes {m, m+16}
        float mnew  = fmaxf(mrun, mt);
        float scale = __expf(mrun - mnew);

        B16U P;
        float psum = 0.f;
#pragma unroll
        for (int i = 0; i < 16; ++i) {
            float p = (av[i] > 0.f) ? __expf(sv[i] - mnew) : 0.f;
            psum += p;
            P.u[i] = f2bf(p);
        }
        psum += __shfl_xor(psum, 16, 32);
        lrun = lrun * scale + psum;
        mrun = mnew;

#pragma unroll
        for (int r = 0; r < 8; ++r) {
            float sr = __shfl(scale, r + hi * 8, 32);
            O0[r] *= sr; O1[r] *= sr; O2[r] *= sr; O3[r] *= sr;
        }

        O0 = wmma_bf16(P.v, Bs0, O0);
        O1 = wmma_bf16(P.v, Bs1, O1);
        O2 = wmma_bf16(P.v, Bs2, O2);
        O3 = wmma_bf16(P.v, Bs3, O3);

        // publish next panel: issuing wave waits TDM, then barrier
        if (w == 0) __builtin_amdgcn_s_wait_tensorcnt(0);
        __syncthreads();
    }

    // -------- epilogue: fold node attention / row sum, add bias ------------
    float crow = na[rowbase + m] / fmaxf(lrun, 1e-30f);
#pragma unroll
    for (int r = 0; r < 8; ++r) {
        float cr = __shfl(crow, r + hi * 8, 32);
        int row = rowbase + r + hi * 8;
#pragma unroll
        for (int c = 0; c < 4; ++c) {
            int col = colbase + c * 16 + m;
            float v = (c == 0) ? O0[r] : (c == 1) ? O1[r] : (c == 2) ? O2[r] : O3[r];
            out[(size_t)row * GD + col] = v * cr + bias[col];
        }
    }
}

// ---------------------------------------------------------------------------
// Launch
// ---------------------------------------------------------------------------
extern "C" void kernel_launch(void* const* d_in, const int* in_sizes, int n_in,
                              void* d_out, int out_size, void* d_ws, size_t ws_size,
                              hipStream_t stream) {
    (void)in_sizes; (void)n_in; (void)out_size; (void)ws_size;

    const float* x      = (const float*)d_in[0];   // [8192, 512]
    const float* adj    = (const float*)d_in[1];   // [8192, 8192]
    const float* weight = (const float*)d_in[2];   // [512, 512]
    const float* bias   = (const float*)d_in[3];   // [512]
    const float* node_w = (const float*)d_in[4];   // [512, 1]
    const float* edge_w = (const float*)d_in[5];   // [512, 512]
    float* out = (float*)d_out;                    // [8192, 512]

    char* ws = (char*)d_ws;
    const size_t NS_OFF  = 0;                        // 8192 f32
    const size_t NA_OFF  = NS_OFF  + 32768;          // 8192 f32
    const size_t XBF_OFF = NA_OFF  + 32768;          // 8192*512 bf16 = 8 MB
    const size_t EWT_OFF = XBF_OFF + 8388608;        // 512*512 bf16
    const size_t WT_OFF  = EWT_OFF + 524288;         // 512*512 bf16
    const size_t H_OFF   = WT_OFF  + 524288;         // 8192*512 bf16 = 8 MB
    const size_t ST_OFF  = H_OFF   + 8388608;        // 512*8192 bf16 = 8 MB (suppT)

    float*          ns    = (float*)(ws + NS_OFF);
    float*          na    = (float*)(ws + NA_OFF);
    unsigned short* xbf   = (unsigned short*)(ws + XBF_OFF);
    unsigned short* ewT   = (unsigned short*)(ws + EWT_OFF);
    unsigned short* wT    = (unsigned short*)(ws + WT_OFF);
    unsigned short* hbuf  = (unsigned short*)(ws + H_OFF);
    unsigned short* suppT = (unsigned short*)(ws + ST_OFF);

    // 1) conversions / transposes
    k_cvt_bf16<<<(GN * GD / 4 + 255) / 256, 256, 0, stream>>>(x, xbf, GN * GD / 4);
    k_transpose_cvt<<<(GD * GD) / 256, 256, 0, stream>>>(edge_w, ewT);
    k_transpose_cvt<<<(GD * GD) / 256, 256, 0, stream>>>(weight, wT);

    // 2) h = x @ edge_w (bf16 row-major), suppT = (x @ weight)^T (bf16)
    k_gemm_xw<<<GN / 16, 256, 0, stream>>>(xbf, ewT, hbuf, 0);
    k_gemm_xw<<<GN / 16, 256, 0, stream>>>(xbf, wT, suppT, 1);

    // 3) node attention softmax (over all nodes)
    k_node_scores<<<GN / 8, 256, 0, stream>>>(x, node_w, ns);
    k_node_softmax<<<1, 256, 0, stream>>>(ns, na);

    // 4) fused masked softmax + SpMM (dynamic LDS: panels + reduce buffers)
    k_flash<<<GN / 16, 256, FLASH_LDS_BYTES, stream>>>(adj, hbuf, suppT, na, bias, out);
}
